// MAMFGCN_48275432407566
// MI455X (gfx1250) — matrix operand order repaired and checked by reading, not compiled, not run on hardware
//
#include <hip/hip_runtime.h>
#include <hip/hip_bf16.h>
#include <math.h>

// ---------------- types for WMMA ----------------
typedef __bf16 bf16_t;
typedef __attribute__((ext_vector_type(16))) __bf16 v16bf;
typedef __attribute__((ext_vector_type(8)))  __bf16 v8bf;
typedef __attribute__((ext_vector_type(8)))  float   v8f;

#define NNODE 3000
#define NPAD  3008          // 3000 padded to multiple of 32
#define NFEATD 256
#define NHIDD  64
#define NLAY   9
#define FEATW  832          // 256 + 9*64
#define OUTD   64

// ======================================================================
// Conversion kernels
// ======================================================================

// fp32 [rows,cols] -> bf16 [rows,ldd] with zero padding in cols [cols,ldd)
__global__ void cvt_adj_bf16(const float* __restrict__ src, bf16_t* __restrict__ dst,
                             int rows, int cols, int ldd) {
    int i = blockIdx.x * 256 + threadIdx.x;
    int total = rows * ldd;
    if (i >= total) return;
    int r = i / ldd, c = i - r * ldd;
    float v = (c < cols) ? src[(size_t)r * cols + c] : 0.0f;
    dst[i] = (bf16_t)v;
}

// x fp32 [3000,256] -> feats bf16 columns [0,256) with row stride FEATW
__global__ void cvt_x_feats(const float* __restrict__ x, bf16_t* __restrict__ feats) {
    int i = blockIdx.x * 256 + threadIdx.x;       // exactly 3000*256 threads
    int r = i >> 8, c = i & 255;
    feats[(size_t)r * FEATW + c] = (bf16_t)x[i];
}

// W fp32 [Kw,64] -> Wt bf16 [64,Kw] (transposed)
__global__ void cvt_w_transpose(const float* __restrict__ W, bf16_t* __restrict__ Wt, int Kw) {
    int i = blockIdx.x * 256 + threadIdx.x;       // exactly Kw*64 threads
    int k = i >> 6, c = i & 63;
    Wt[(size_t)c * Kw + k] = (bf16_t)W[i];
}

// ======================================================================
// Fragment loaders (lane layouts per CDNA5 ISA 7.12.2)
//   A 16x32 bf16: lane(l<16) M=l, K=kbA+{0..7},{16..23}; l>=16: K base +8
//   B 32x16 bf16 (from transposed Bt): lane col = l&15, K same as A: {0..7},{16..23}; l>=16: K base +8
// ======================================================================
__device__ __forceinline__ v16bf ldA(const bf16_t* p) {
    v8bf lo = *(const v8bf*)p;          // K = +0..7
    v8bf hi = *(const v8bf*)(p + 16);   // K = +16..23
    v16bf a;
#pragma unroll
    for (int e = 0; e < 8; ++e) { a[e] = lo[e]; a[e + 8] = hi[e]; }
    return a;
}
__device__ __forceinline__ v16bf ldB(const bf16_t* p) {
    v8bf lo = *(const v8bf*)p;          // K = +0..7
    v8bf hi = *(const v8bf*)(p + 16);   // K = +16..23
    v16bf b;
#pragma unroll
    for (int e = 0; e < 8; ++e) { b[e] = lo[e]; b[e + 8] = hi[e]; }
    return b;
}

// ======================================================================
// WMMA GEMM: C[M,64] = A[M,Kpad](bf16,row-major,lda) x Bt[64,Kpad](bf16, B^T)
//   Kpad: multiple of 32 with an EVEN number of 32-steps (true for all calls);
//         padded regions of A/Bt contain zeros.
//   mode 0: outB[col*ldo + row] = bf16(acc)         (transposed panel, rows>=M -> 0)
//   mode 1: outB[row*ldo + col_off + col] = bf16(tanh(acc + bias[col]))
//   mode 2: outF[row*64 + col]  = acc + bias[col]
// Wave = 16 rows x 64 cols (4 wmma tiles), register double-buffered K loop.
// Block = 2 waves. Grid = NPAD/32 = 94.
// ======================================================================
__global__ __launch_bounds__(64)
void wmma_gemm_n64(const bf16_t* __restrict__ A, int lda,
                   const bf16_t* __restrict__ Bt, int ldbt,
                   int M, int Kpad, int mode,
                   const float* __restrict__ bias,
                   bf16_t* __restrict__ outB, int ldo, int col_off,
                   float* __restrict__ outF) {
    const int lane = threadIdx.x & 31;
    const int wv   = threadIdx.x >> 5;
    const int row0 = (blockIdx.x * 2 + wv) * 16;
    const int nloc = lane & 15;
    const int hi   = lane >> 4;

    int arow = row0 + nloc;                 // this lane supplies A row (M = lane&15)
    if (arow > M - 1) arow = M - 1;         // clamp; results masked on store
    const bf16_t* __restrict__ aBase = A  + (size_t)arow * lda + hi * 8;
    const bf16_t* __restrict__ bBase = Bt + (size_t)nloc * ldbt + hi * 8;
    const size_t bts = (size_t)ldbt * 16;   // stride between 16-col B tiles

    v8f acc0 = {}, acc1 = {}, acc2 = {}, acc3 = {};
    v16bf aX, X0, X1, X2, X3;               // ping buffer
    v16bf aY, Y0, Y1, Y2, Y3;               // pong buffer

#define LOADSET(S, k)                                                     \
    do {                                                                  \
        a##S = ldA(aBase + (k));                                          \
        S##0 = ldB(bBase + (k));                                          \
        S##1 = ldB(bBase + bts + (k));                                    \
        S##2 = ldB(bBase + 2 * bts + (k));                                \
        S##3 = ldB(bBase + 3 * bts + (k));                                \
    } while (0)

#define WMMASET(S)                                                        \
    do {                                                                  \
        acc0 = __builtin_amdgcn_wmma_f32_16x16x32_bf16(                   \
            false, a##S, false, S##0, (short)0, acc0, false, false);      \
        acc1 = __builtin_amdgcn_wmma_f32_16x16x32_bf16(                   \
            false, a##S, false, S##1, (short)0, acc1, false, false);      \
        acc2 = __builtin_amdgcn_wmma_f32_16x16x32_bf16(                   \
            false, a##S, false, S##2, (short)0, acc2, false, false);      \
        acc3 = __builtin_amdgcn_wmma_f32_16x16x32_bf16(                   \
            false, a##S, false, S##3, (short)0, acc3, false, false);      \
    } while (0)

    const int nIter = Kpad >> 5;            // even, >= 8 for all call sites
    LOADSET(X, 0);
    LOADSET(Y, 32);
    for (int it = 0; it < nIter - 2; it += 2) {
        WMMASET(X);
        LOADSET(X, (size_t)(it + 2) * 32);  // issued in the shadow of the WMMAs
        WMMASET(Y);
        LOADSET(Y, (size_t)(it + 3) * 32);
        if (it + 10 < nIter)
            __builtin_prefetch(aBase + (size_t)(it + 10) * 32, 0, 3);
    }
    WMMASET(X);
    WMMASET(Y);
#undef LOADSET
#undef WMMASET

    // ---- epilogue (C/D layout: VGPR r -> M = r + (lane>>4)*8, N = lane&15) ----
    v8f accA[4] = {acc0, acc1, acc2, acc3};
    const int rbase = hi * 8;
    if (mode == 0) {
#pragma unroll
        for (int j = 0; j < 4; ++j) {
            int col = j * 16 + nloc;
            v8bf ov;
#pragma unroll
            for (int r = 0; r < 8; ++r) {
                int row = row0 + rbase + r;
                float v = (row < M) ? accA[j][r] : 0.0f;  // zero K-padding of panel
                ov[r] = (bf16_t)v;
            }
            // 8 consecutive rows of Pt are contiguous -> one 16B store
            *(v8bf*)(outB + (size_t)col * ldo + row0 + rbase) = ov;
        }
    } else if (mode == 1) {
#pragma unroll
        for (int j = 0; j < 4; ++j) {
            int col = j * 16 + nloc;
            float bv = bias[col];
#pragma unroll
            for (int r = 0; r < 8; ++r) {
                int row = row0 + rbase + r;
                if (row < M)
                    outB[(size_t)row * ldo + col_off + col] =
                        (bf16_t)tanhf(accA[j][r] + bv);
            }
        }
    } else {
#pragma unroll
        for (int j = 0; j < 4; ++j) {
            int col = j * 16 + nloc;
            float bv = bias[col];
#pragma unroll
            for (int r = 0; r < 8; ++r) {
                int row = row0 + rbase + r;
                if (row < M)
                    outF[(size_t)row * 64 + col] = accA[j][r] + bv;
            }
        }
    }
}

// ======================================================================
// Row-wise L2 normalize: out[row,:] = in[row,:] / max(||in[row,:]||, 1e-12)
// ======================================================================
__global__ void rownorm64(const float* __restrict__ I, float* __restrict__ O) {
    __shared__ float red[64];
    int row = blockIdx.x, t = threadIdx.x;
    float v = I[(size_t)row * 64 + t];
    red[t] = v * v;
    __syncthreads();
    for (int s = 32; s > 0; s >>= 1) {
        if (t < s) red[t] += red[t + s];
        __syncthreads();
    }
    float nrm = fmaxf(sqrtf(red[0]), 1e-12f);
    O[(size_t)row * 64 + t] = v / nrm;
}

// ======================================================================
// Attention over 4 views + softmax + MLP head (per node)
// ======================================================================
__global__ void att_mlp(const float* __restrict__ e1, const float* __restrict__ e2,
                        const float* __restrict__ e3, const float* __restrict__ c1,
                        const float* __restrict__ c2, const float* __restrict__ c3,
                        const float* __restrict__ aw1, const float* __restrict__ ab1,
                        const float* __restrict__ aw2, const float* __restrict__ mw,
                        const float* __restrict__ mb,
                        float* __restrict__ outp, float* __restrict__ betao) {
    int n = blockIdx.x * blockDim.x + threadIdx.x;
    if (n >= NNODE) return;
    const float* Z[3] = {e1, e2, e3};
    float wlog[4];
    for (int v = 0; v < 4; ++v) {
        float s0 = 0.f, s1 = 0.f;
        for (int d = 0; d < 64; ++d) {
            float zv = (v < 3) ? Z[v][(size_t)n * 64 + d]
                               : (c1[(size_t)n * 64 + d] + c2[(size_t)n * 64 + d] +
                                  c3[(size_t)n * 64 + d]) * (1.0f / 3.0f);
            s0 += zv * aw1[d * 2 + 0];
            s1 += zv * aw1[d * 2 + 1];
        }
        wlog[v] = tanhf(s0 + ab1[0]) * aw2[0] + tanhf(s1 + ab1[1]) * aw2[1];
    }
    float mx = fmaxf(fmaxf(wlog[0], wlog[1]), fmaxf(wlog[2], wlog[3]));
    float beta[4], bsum = 0.f;
    for (int v = 0; v < 4; ++v) { beta[v] = expf(wlog[v] - mx); bsum += beta[v]; }
    for (int v = 0; v < 4; ++v) { beta[v] /= bsum; betao[(size_t)n * 4 + v] = beta[v]; }
    float l0 = 0.f, l1 = 0.f;
    for (int d = 0; d < 64; ++d) {
        float xc = (c1[(size_t)n * 64 + d] + c2[(size_t)n * 64 + d] +
                    c3[(size_t)n * 64 + d]) * (1.0f / 3.0f);
        float e = beta[0] * e1[(size_t)n * 64 + d] + beta[1] * e2[(size_t)n * 64 + d] +
                  beta[2] * e3[(size_t)n * 64 + d] + beta[3] * xc;
        l0 += e * mw[d * 2 + 0];
        l1 += e * mw[d * 2 + 1];
    }
    l0 += mb[0]; l1 += mb[1];
    float m2 = fmaxf(l0, l1);
    float p0 = expf(l0 - m2), p1 = expf(l1 - m2), ps = p0 + p1;
    outp[(size_t)n * 2 + 0] = p0 / ps;
    outp[(size_t)n * 2 + 1] = p1 / ps;
}

// ======================================================================
// Host orchestration
// ======================================================================
struct SnowParams {
    const float* Ws[NLAY];
    const float* bs[NLAY];
    const float* Wo;
    const float* bo;
};

extern "C" void kernel_launch(void* const* d_in, const int* in_sizes, int n_in,
                              void* d_out, int out_size, void* d_ws, size_t ws_size,
                              hipStream_t stream) {
    (void)n_in; (void)out_size; (void)ws_size;

    const float* x     = (const float*)d_in[0];
    const float* sadj  = (const float*)d_in[1];
    const float* fadj  = (const float*)d_in[2];
    const float* fadj2 = (const float*)d_in[3];

    // Param groups of 20 arrays each start at index 4. Detect flatten order:
    //  - jax tree (alphabetical keys): Wo, Ws[0..8], bo, bs[0..8]  (Wo size 53248 first)
    //  - dict insertion order:         Ws[0..8], bs[0..8], Wo, bo
    auto getSnow = [&](int g) -> SnowParams {
        SnowParams s;
        if (in_sizes[g] == FEATW * 64) {   // alphabetical: Wo first
            s.Wo = (const float*)d_in[g + 0];
            for (int i = 0; i < NLAY; ++i) s.Ws[i] = (const float*)d_in[g + 1 + i];
            s.bo = (const float*)d_in[g + 10];
            for (int i = 0; i < NLAY; ++i) s.bs[i] = (const float*)d_in[g + 11 + i];
        } else {                           // insertion order
            for (int i = 0; i < NLAY; ++i) s.Ws[i] = (const float*)d_in[g + i];
            for (int i = 0; i < NLAY; ++i) s.bs[i] = (const float*)d_in[g + 9 + i];
            s.Wo = (const float*)d_in[g + 18];
            s.bo = (const float*)d_in[g + 19];
        }
        return s;
    };
    SnowParams sg1 = getSnow(4), sg2 = getSnow(24), sg3 = getSnow(44), cg = getSnow(64);
    const float* att_w1 = (const float*)d_in[84];
    const float* att_b1 = (const float*)d_in[85];
    const float* att_w2 = (const float*)d_in[86];
    const float* mlp_w  = (const float*)d_in[87];
    const float* mlp_b  = (const float*)d_in[88];

    // ---- workspace layout (byte offsets, all 16B aligned) ----
    char* ws = (char*)d_ws;
    const size_t ADJ_BYTES = (size_t)NNODE * NPAD * 2;          // 18,048,000
    bf16_t* adjS  = (bf16_t*)(ws + 0);
    bf16_t* adjF  = (bf16_t*)(ws + ADJ_BYTES);
    bf16_t* adjF2 = (bf16_t*)(ws + 2 * ADJ_BYTES);
    bf16_t* feats = (bf16_t*)(ws + 3 * ADJ_BYTES);              // [3000, 832]
    const size_t FEATS_BYTES = (size_t)NNODE * FEATW * 2;       // 4,992,000
    bf16_t* Wt    = (bf16_t*)(ws + 3 * ADJ_BYTES + FEATS_BYTES);// 348,160 elems
    const size_t WT_BYTES = (size_t)348160 * 2;
    bf16_t* Pt    = (bf16_t*)(ws + 3 * ADJ_BYTES + FEATS_BYTES + WT_BYTES); // [64,3008]
    const size_t PT_BYTES = (size_t)64 * NPAD * 2;
    float*  Of    = (float*)(ws + 3 * ADJ_BYTES + FEATS_BYTES + WT_BYTES + PT_BYTES);

    // ---- output layout: output, beta, emb1, com1, com2, com3, emb2, emb3 ----
    float* out   = (float*)d_out;
    float* outp  = out + 0;
    float* betao = out + 6000;
    float* emb1  = out + 18000;
    float* com1  = out + 18000 + 1 * 192000;
    float* com2  = out + 18000 + 2 * 192000;
    float* com3  = out + 18000 + 3 * 192000;
    float* emb2  = out + 18000 + 4 * 192000;
    float* emb3  = out + 18000 + 5 * 192000;

    // ---- convert adjacencies to padded bf16 (L2-resident working set) ----
    const int adjBlocks = (NNODE * NPAD) / 256;   // 35,250 exact
    cvt_adj_bf16<<<adjBlocks, 256, 0, stream>>>(sadj,  adjS,  NNODE, NNODE, NPAD);
    cvt_adj_bf16<<<adjBlocks, 256, 0, stream>>>(fadj,  adjF,  NNODE, NNODE, NPAD);
    cvt_adj_bf16<<<adjBlocks, 256, 0, stream>>>(fadj2, adjF2, NNODE, NNODE, NPAD);

    auto run_snowball = [&](const SnowParams& p, const bf16_t* adj, float* emb_out) {
        // feats[:, 0:256] = bf16(x)
        cvt_x_feats<<<(NNODE * NFEATD) / 256, 256, 0, stream>>>(x, feats);
        // weights -> transposed bf16
        size_t woff = 0;
        for (int i = 0; i < NLAY; ++i) {
            int Kw = NFEATD + NHIDD * i;
            cvt_w_transpose<<<(Kw * 64) / 256, 256, 0, stream>>>(p.Ws[i], Wt + woff, Kw);
            woff += (size_t)Kw * 64;
        }
        cvt_w_transpose<<<(FEATW * 64) / 256, 256, 0, stream>>>(p.Wo, Wt + woff, FEATW);
        // layers
        size_t w2 = 0;
        for (int i = 0; i < NLAY; ++i) {
            int Kw = NFEATD + NHIDD * i;   // multiple of 64 -> even # of 32-steps
            // Pt = (feats[:, :Kw] @ W_i)^T
            wmma_gemm_n64<<<NPAD / 32, 64, 0, stream>>>(
                feats, FEATW, Wt + w2, Kw, NNODE, Kw, 0, nullptr, Pt, NPAD, 0, nullptr);
            // feats[:, 256+64i : 256+64(i+1)] = tanh(adj @ P + b_i)
            wmma_gemm_n64<<<NPAD / 32, 64, 0, stream>>>(
                adj, NPAD, Pt, NPAD, NNODE, NPAD, 1, p.bs[i], feats, FEATW,
                NFEATD + NHIDD * i, nullptr);
            w2 += (size_t)Kw * 64;
        }
        // Pt = (feats @ Wo)^T ; O = adj @ P + bo ; emb = rownorm(O)
        wmma_gemm_n64<<<NPAD / 32, 64, 0, stream>>>(
            feats, FEATW, Wt + w2, FEATW, NNODE, FEATW, 0, nullptr, Pt, NPAD, 0, nullptr);
        wmma_gemm_n64<<<NPAD / 32, 64, 0, stream>>>(
            adj, NPAD, Pt, NPAD, NNODE, NPAD, 2, p.bo, nullptr, 0, 0, Of);
        rownorm64<<<NNODE, 64, 0, stream>>>(Of, emb_out);
    };

    run_snowball(sg1, adjS,  emb1);
    run_snowball(sg2, adjF,  emb2);
    run_snowball(sg3, adjF2, emb3);
    run_snowball(cg,  adjS,  com1);
    run_snowball(cg,  adjF,  com2);
    run_snowball(cg,  adjF2, com3);

    att_mlp<<<(NNODE + 127) / 128, 128, 0, stream>>>(
        emb1, emb2, emb3, com1, com2, com3,
        att_w1, att_b1, att_w2, mlp_w, mlp_b, outp, betao);
}
